// MultiHeadSelfAttention_68564857913384
// MI455X (gfx1250) — compile-verified, matrix operand order checked by reading, not run on hardware
//
#include <hip/hip_runtime.h>

typedef __attribute__((ext_vector_type(16))) __bf16 v16bf;
typedef __attribute__((ext_vector_type(8)))  float  v8f;
typedef __attribute__((ext_vector_type(4)))  unsigned v4u;

#define DIMC   512
#define NHEAD  8
#define HDIM   64
#define NTOK   1024
#define NBATCH 8
#define ATTN_SCALE 0.125f   // 64^-0.5

// ---------- bf16 helpers ----------
static __device__ __forceinline__ unsigned short f2bfu(float f) {
  unsigned u = __builtin_bit_cast(unsigned, f);
  unsigned r = u + 0x7FFFu + ((u >> 16) & 1u);   // round-to-nearest-even
  return (unsigned short)(r >> 16);
}

static __device__ __forceinline__ v8f wmma_bf16(v16bf a, v16bf b, v8f c) {
  return __builtin_amdgcn_wmma_f32_16x16x32_bf16(
      /*neg_a=*/false, a, /*neg_b=*/false, b,
      /*c_mod=*/(short)0, c, /*reuse_a=*/false, /*reuse_b=*/false);
}

// 16-bit A/B fragment from a K-contiguous row: lane's 16 bf16 values are two
// aligned 16B chunks at k = 8*half and k = 16+8*half (ISA 16x32 layout).
struct frag2 { v4u lo, hi; };
static __device__ __forceinline__ v16bf load_frag_row(const unsigned short* rowp, int half) {
  frag2 f;
  f.lo = *(const v4u*)(rowp + (half << 3));
  f.hi = *(const v4u*)(rowp + 16 + (half << 3));
  return __builtin_bit_cast(v16bf, f);
}

// =====================================================================
// Prep: src[z][K][N] fp32 -> dst[z][N][K] bf16 (transpose + convert),
// LDS-tiled so both global sides are coalesced. Tile = 32(K) x 64(N).
// =====================================================================
__global__ __launch_bounds__(256) void transpose_cvt_kernel(
    const float* __restrict__ src, unsigned short* __restrict__ dst, int K, int N) {
  __shared__ alignas(16) unsigned short tile[64][40];  // [n][k]
  src += (size_t)blockIdx.z * K * N;
  dst += (size_t)blockIdx.z * K * N;
  const int k0 = blockIdx.y * 32, n0 = blockIdx.x * 64;
  const int t = threadIdx.x;
#pragma unroll
  for (int i = 0; i < 8; ++i) {
    int idx = t + i * 256;
    int n = idx & 63, k = idx >> 6;
    tile[n][k] = f2bfu(src[(size_t)(k0 + k) * N + n0 + n]);
  }
  __syncthreads();
#pragma unroll
  for (int i = 0; i < 8; ++i) {
    int idx = t + i * 256;
    int k = idx & 31, n = idx >> 5;
    dst[(size_t)(n0 + n) * K + k0 + k] = tile[n][k];
  }
}

// =====================================================================
// Kernel 1: qkv = xt @ wTq^T + b_qkv -> bf16 qkv_ws[s][b][h][n][d]
//   Block 256 thr = 8 waves (4m x 2n), 128x128 tile; wave tile 32x64
//   (2x4 WMMA, 0.75 frag loads per WMMA). No LDS: fragments are direct
//   global b128 loads, WGP$/L2 serve the reuse.
// =====================================================================
__global__ __launch_bounds__(256) void qkv_gemm_kernel(
    const unsigned short* __restrict__ xt, const unsigned short* __restrict__ wTq,
    const float* __restrict__ b_qkv, unsigned short* __restrict__ qkv_ws) {
  const int t = threadIdx.x;
  const int lane = t & 31, half = lane >> 4, nlane = lane & 15;
  const int w = t >> 5, wm = w >> 1, wn = w & 1;   // 4x2 wave grid
  const int row0 = blockIdx.y * 128;
  const int n0 = blockIdx.x * 128;

  const unsigned short* arow[2];
  const unsigned short* brow[4];
#pragma unroll
  for (int mi = 0; mi < 2; ++mi)
    arow[mi] = xt + (size_t)(row0 + wm * 32 + mi * 16 + nlane) * DIMC;
#pragma unroll
  for (int ni = 0; ni < 4; ++ni)
    brow[ni] = wTq + (size_t)(n0 + wn * 64 + ni * 16 + nlane) * DIMC;

  v8f acc[2][4] = {};
  for (int k0 = 0; k0 < DIMC; k0 += 32) {
    v16bf af[2], bf[4];
#pragma unroll
    for (int mi = 0; mi < 2; ++mi) af[mi] = load_frag_row(arow[mi] + k0, half);
#pragma unroll
    for (int ni = 0; ni < 4; ++ni) bf[ni] = load_frag_row(brow[ni] + k0, half);
#pragma unroll
    for (int mi = 0; mi < 2; ++mi)
#pragma unroll
      for (int ni = 0; ni < 4; ++ni)
        acc[mi][ni] = wmma_bf16(af[mi], bf[ni], acc[mi][ni]);
  }

#pragma unroll
  for (int mi = 0; mi < 2; ++mi)
#pragma unroll
    for (int ni = 0; ni < 4; ++ni) {
      int col = n0 + wn * 64 + ni * 16 + nlane;
      float bias = b_qkv[col];
      int s = col >> 9, rem = col & 511;
      int hh = rem >> 6, dd = rem & 63;
#pragma unroll
      for (int r = 0; r < 8; ++r) {
        int grow = row0 + wm * 32 + mi * 16 + r + 8 * half;
        int bb = grow >> 10, nn = grow & 1023;
        size_t o = ((((size_t)s * NBATCH + bb) * NHEAD + hh) * NTOK + nn) * HDIM + dd;
        qkv_ws[o] = f2bfu(acc[mi][ni][r] + bias);
      }
    }
}

// =====================================================================
// Kernel 2: flash attention. Block = 128 q-rows of one (b,h); 8 waves,
// each wave owns 16 q-rows; KV streamed in 32-wide tiles.
// =====================================================================
__global__ __launch_bounds__(256) void attn_kernel(
    const unsigned short* __restrict__ qkv_ws, unsigned short* __restrict__ attn_ws) {
  __shared__ alignas(16) unsigned short Ps[8][16][40];  // per-wave P (16q x 32kv)
  __shared__ alignas(16) unsigned short Vt[64][40];     // V^T tile: [d][kv]

  const int t = threadIdx.x;
  const int lane = t & 31, half = lane >> 4, nlane = lane & 15;
  const int w = t >> 5;
  const int bh = blockIdx.y;
  const int b = bh >> 3, h = bh & 7;
  const int qr0 = blockIdx.x * 128 + w * 16;

  const size_t head_off = ((size_t)b * NHEAD + h) * NTOK * HDIM;
  const size_t plane = (size_t)NBATCH * NHEAD * NTOK * HDIM;
  const unsigned short* qp = qkv_ws + head_off;
  const unsigned short* kp = qkv_ws + plane + head_off;
  const unsigned short* vp = qkv_ws + 2 * plane + head_off;

  v16bf qa[2];
#pragma unroll
  for (int ci = 0; ci < 2; ++ci)
    qa[ci] = load_frag_row(qp + (size_t)(qr0 + nlane) * HDIM + ci * 32, half);

  float rmax[8], rsum[8];
#pragma unroll
  for (int r = 0; r < 8; ++r) { rmax[r] = -1e30f; rsum[r] = 0.f; }
  v8f oacc[4] = {};

  for (int kv0 = 0; kv0 < NTOK; kv0 += 32) {
    __syncthreads();                         // protect Vt vs previous reads
#pragma unroll
    for (int i = 0; i < 8; ++i) {            // stage V^T: [d][kv]
      int idx = t + i * 256;
      int d = idx & 63, kv = idx >> 6;
      Vt[d][kv] = vp[(size_t)(kv0 + kv) * HDIM + d];
    }
    __syncthreads();

    // S = Q @ K^T : two 16x16 tiles, contraction over D=64 in 2 chunks.
    v8f s0 = {}, s1 = {};
#pragma unroll
    for (int ci = 0; ci < 2; ++ci) {
      v16bf kb0 = load_frag_row(kp + (size_t)(kv0 + nlane) * HDIM + ci * 32, half);
      v16bf kb1 = load_frag_row(kp + (size_t)(kv0 + 16 + nlane) * HDIM + ci * 32, half);
      s0 = wmma_bf16(qa[ci], kb0, s0);
      s1 = wmma_bf16(qa[ci], kb1, s1);
    }

    // Online softmax per q-row (row = r + 8*half, spread over 16 lanes).
#pragma unroll
    for (int r = 0; r < 8; ++r) {
      float a0 = s0[r] * ATTN_SCALE;
      float a1 = s1[r] * ATTN_SCALE;
      float tmax = fmaxf(a0, a1);
#pragma unroll
      for (int m = 1; m < 16; m <<= 1)
        tmax = fmaxf(tmax, __shfl_xor(tmax, m, 32));
      float mnew = fmaxf(rmax[r], tmax);
      float corr = __expf(rmax[r] - mnew);
      rmax[r] = mnew;
      float p0 = __expf(a0 - mnew);
      float p1 = __expf(a1 - mnew);
      float psum = p0 + p1;
#pragma unroll
      for (int m = 1; m < 16; m <<= 1)
        psum += __shfl_xor(psum, m, 32);
      rsum[r] = rsum[r] * corr + psum;
#pragma unroll
      for (int ci = 0; ci < 4; ++ci) oacc[ci][r] *= corr;
      int mrow = r + 8 * half;
      Ps[w][mrow][nlane]      = f2bfu(p0);
      Ps[w][mrow][16 + nlane] = f2bfu(p1);
    }
    asm volatile("s_wait_dscnt 0" ::: "memory");  // P stores -> fragment reload

    // O += P @ V
    v16bf pa = load_frag_row(&Ps[w][nlane][0], half);
#pragma unroll
    for (int ci = 0; ci < 4; ++ci) {
      v16bf vb = load_frag_row(&Vt[ci * 16 + nlane][0], half);
      oacc[ci] = wmma_bf16(pa, vb, oacc[ci]);
    }
  }

  // Normalize, write attn_ws[(b*1024+n)][h*64+d] (bf16 8192x512 row-major).
#pragma unroll
  for (int r = 0; r < 8; ++r) {
    float inv = 1.0f / rsum[r];
    int qrow = qr0 + r + 8 * half;
#pragma unroll
    for (int ci = 0; ci < 4; ++ci) {
      attn_ws[((size_t)b * NTOK + qrow) * DIMC + h * HDIM + ci * 16 + nlane] =
          f2bfu(oacc[ci][r] * inv);
    }
  }
}

// =====================================================================
// Kernel 3: out = attn @ wTp^T + b_proj, stored transposed (B,C,N) fp32.
// Same 128x128 block / 32x64 wave tiling, no LDS.
// =====================================================================
__global__ __launch_bounds__(256) void proj_gemm_kernel(
    const unsigned short* __restrict__ attn_ws, const unsigned short* __restrict__ wTp,
    const float* __restrict__ b_proj, float* __restrict__ out) {
  const int t = threadIdx.x;
  const int lane = t & 31, half = lane >> 4, nlane = lane & 15;
  const int w = t >> 5, wm = w >> 1, wn = w & 1;
  const int row0 = blockIdx.y * 128;
  const int n0 = blockIdx.x * 128;

  const unsigned short* arow[2];
  const unsigned short* brow[4];
#pragma unroll
  for (int mi = 0; mi < 2; ++mi)
    arow[mi] = attn_ws + (size_t)(row0 + wm * 32 + mi * 16 + nlane) * DIMC;
#pragma unroll
  for (int ni = 0; ni < 4; ++ni)
    brow[ni] = wTp + (size_t)(n0 + wn * 64 + ni * 16 + nlane) * DIMC;

  v8f acc[2][4] = {};
  for (int k0 = 0; k0 < DIMC; k0 += 32) {
    v16bf af[2], bf[4];
#pragma unroll
    for (int mi = 0; mi < 2; ++mi) af[mi] = load_frag_row(arow[mi] + k0, half);
#pragma unroll
    for (int ni = 0; ni < 4; ++ni) bf[ni] = load_frag_row(brow[ni] + k0, half);
#pragma unroll
    for (int mi = 0; mi < 2; ++mi)
#pragma unroll
      for (int ni = 0; ni < 4; ++ni)
        acc[mi][ni] = wmma_bf16(af[mi], bf[ni], acc[mi][ni]);
  }

#pragma unroll
  for (int mi = 0; mi < 2; ++mi)
#pragma unroll
    for (int ni = 0; ni < 4; ++ni) {
      int col = n0 + wn * 64 + ni * 16 + nlane;
      float bias = b_proj[col];
#pragma unroll
      for (int r = 0; r < 8; ++r) {
        int grow = row0 + wm * 32 + mi * 16 + r + 8 * half;
        int bb = grow >> 10, nn = grow & 1023;
        out[((size_t)bb * DIMC + col) * NTOK + nn] = acc[mi][ni][r] + bias;
      }
    }
}

// =====================================================================
extern "C" void kernel_launch(void* const* d_in, const int* in_sizes, int n_in,
                              void* d_out, int out_size, void* d_ws, size_t ws_size,
                              hipStream_t stream) {
  const float* x      = (const float*)d_in[0];
  const float* w_qkv  = (const float*)d_in[1];
  const float* b_qkv  = (const float*)d_in[2];
  const float* w_proj = (const float*)d_in[3];
  const float* b_proj = (const float*)d_in[4];
  float* out = (float*)d_out;

  unsigned short* qkv_ws  = (unsigned short*)d_ws;                              // 3*8*8*1024*64
  unsigned short* attn_ws = qkv_ws + (size_t)3 * NBATCH * NHEAD * NTOK * HDIM;  // 8192*512
  unsigned short* xt      = attn_ws + (size_t)NBATCH * NTOK * DIMC;             // 8192*512
  unsigned short* wTq     = xt + (size_t)NBATCH * NTOK * DIMC;                  // 1536*512
  unsigned short* wTp     = wTq + (size_t)3 * DIMC * DIMC;                      // 512*512

  dim3 blk(256);
  // x (B,C,N) -> xt (B,N,C): per-batch 512x1024 transpose+convert.
  transpose_cvt_kernel<<<dim3(NTOK / 64, DIMC / 32, NBATCH), blk, 0, stream>>>(
      x, xt, DIMC, NTOK);
  // w_qkv (512,1536) -> wTq (1536,512); w_proj (512,512) -> wTp (512,512).
  transpose_cvt_kernel<<<dim3(3 * DIMC / 64, DIMC / 32, 1), blk, 0, stream>>>(
      w_qkv, wTq, DIMC, 3 * DIMC);
  transpose_cvt_kernel<<<dim3(DIMC / 64, DIMC / 32, 1), blk, 0, stream>>>(
      w_proj, wTp, DIMC, DIMC);

  qkv_gemm_kernel<<<dim3(12, 64), blk, 0, stream>>>(xt, wTq, b_qkv, qkv_ws);
  attn_kernel<<<dim3(8, 64), blk, 0, stream>>>(qkv_ws, attn_ws);
  proj_gemm_kernel<<<dim3(4, 64), blk, 0, stream>>>(attn_ws, wTp, b_proj, out);
}
